// CrossAttention_30408368456177
// MI455X (gfx1250) — matrix operand order repaired and checked by reading, not compiled, not run on hardware
//
#include <hip/hip_runtime.h>

typedef __attribute__((ext_vector_type(16))) _Float16 v16h;
typedef __attribute__((ext_vector_type(8)))  _Float16 v8h;
typedef __attribute__((ext_vector_type(8)))  float    v8f;
typedef int v4i_ __attribute__((vector_size(16)));

#if defined(__HIP_DEVICE_COMPILE__) &&                                      \
    __has_builtin(__builtin_amdgcn_global_load_async_to_lds_b128) &&        \
    __has_builtin(__builtin_amdgcn_s_wait_asynccnt)
#define HAVE_ASYNC_LDS 1
#else
#define HAVE_ASYNC_LDS 0
#endif

// 16B global -> LDS copy; async (ASYNCcnt) when the gfx1250 builtin exists
static __device__ __forceinline__ void copy16_g2l(const void* g, void* l) {
#if HAVE_ASYNC_LDS
  __builtin_amdgcn_global_load_async_to_lds_b128(
      (__attribute__((address_space(1))) v4i_*)g,
      (__attribute__((address_space(3))) v4i_*)l, 0, 0);
#else
  *(uint4*)l = *(const uint4*)g;
#endif
}
static __device__ __forceinline__ void copy16_wait() {
#if HAVE_ASYNC_LDS
  __builtin_amdgcn_s_wait_asynccnt(0);
#endif
}

static __device__ __forceinline__ v8f wmma16(v16h a, v16h b, v8f c) {
  return __builtin_amdgcn_wmma_f32_16x16x32_f16(false, a, false, b, (short)0, c,
                                                false, false);
}

// ---- fragment loaders (wave32); contiguous k order, same for A and B, not the ISA table order ----
// All loaders below read two contiguous 16B runs per lane -> ds_load_b128.

// A 16x32 (MxK) from row-major [M][stride]: lane row = lane%16,
// elems e -> K = klo+e, klo = (lane>=16)?16:0. Contiguous in k.
static __device__ __forceinline__ v16h load_a_frag(const _Float16* base, int stride, int lane) {
  const int row = lane & 15;
  const int klo = lane & 16;  // 0 or 16
  const _Float16* p = base + row * stride + klo;
  v16h a;
#pragma unroll
  for (int e = 0; e < 16; ++e) a[e] = p[e];
  return a;
}

// B 32x16 (KxN) from n-major source (B(k,n) = base[n*stride + k]):
// lane col = lane%16, elems e -> K = e + ((lane>=16)?16:0). Contiguous in k.
static __device__ __forceinline__ v16h load_b_frag_t(const _Float16* base, int stride, int lane) {
  const int col = lane & 15;
  const int k0  = lane & 16;
  const _Float16* p = base + (size_t)col * stride + k0;
  v16h b;
#pragma unroll
  for (int e = 0; e < 16; ++e) b[e] = p[e];
  return b;
}

// ------------------------------- kernels -------------------------------

__global__ void cvt_f32_f16_k(const float* __restrict__ src, _Float16* __restrict__ dst, int n) {
  int i = blockIdx.x * blockDim.x + threadIdx.x;
  if (i < n) dst[i] = (_Float16)src[i];
}

// LayerNorm over 512-wide rows, one wave per row, output f16
__global__ __launch_bounds__(256) void ln512_f16_k(const float* __restrict__ x,
                                                   const float* __restrict__ gamma,
                                                   const float* __restrict__ beta,
                                                   _Float16* __restrict__ y, int rows) {
  const int lane = threadIdx.x & 31;
  const int wave = threadIdx.x >> 5;
  const int row  = blockIdx.x * 8 + wave;
  if (row >= rows) return;
  const float* xr = x + (size_t)row * 512;
  float4 v[4];
  float s = 0.f;
#pragma unroll
  for (int i = 0; i < 4; ++i) {
    v[i] = ((const float4*)xr)[lane + i * 32];
    s += v[i].x + v[i].y + v[i].z + v[i].w;
  }
#pragma unroll
  for (int m = 16; m >= 1; m >>= 1) s += __shfl_xor(s, m, 32);
  const float mu = s * (1.f / 512.f);
  float var = 0.f;
#pragma unroll
  for (int i = 0; i < 4; ++i) {
    float a = v[i].x - mu, b = v[i].y - mu, c = v[i].z - mu, d = v[i].w - mu;
    var += a * a + b * b + c * c + d * d;
  }
#pragma unroll
  for (int m = 16; m >= 1; m >>= 1) var += __shfl_xor(var, m, 32);
  const float rstd = rsqrtf(var * (1.f / 512.f) + 1e-5f);
  _Float16* yr = y + (size_t)row * 512;
#pragma unroll
  for (int i = 0; i < 4; ++i) {
    const int base = (lane + i * 32) * 4;
    float e[4] = {v[i].x, v[i].y, v[i].z, v[i].w};
#pragma unroll
    for (int j = 0; j < 4; ++j)
      yr[base + j] = (_Float16)((e[j] - mu) * rstd * gamma[base + j] + beta[base + j]);
  }
}

// C[M,N](f32,+bias) = A[M,K]f16 @ B[K,N]f16 ; 128x128 block tile, BK=32.
// 8 waves in 4(m) x 2(n); wave tile 32x64 -> 2x4 WMMA frags, 8 wmma / K-step.
// B is staged TRANSPOSED in LDS so every frag load is contiguous b128.
__global__ __launch_bounds__(256) void gemm_f16_k(const _Float16* __restrict__ A,
                                                  const _Float16* __restrict__ B,
                                                  const float* __restrict__ bias,
                                                  float* __restrict__ Cf,
                                                  _Float16* __restrict__ Ch,
                                                  int M, int N, int K) {
  __shared__ _Float16 As[128][40];  // row-major, 80B row stride
  __shared__ _Float16 Bt[128][40];  // Bt[n][k]
  const int tid  = threadIdx.x;
  const int lane = tid & 31;
  const int wave = tid >> 5;
  const int wm   = wave & 3;   // 4 row groups of 32
  const int wn   = wave >> 2;  // 2 col groups of 64
  const int m0   = blockIdx.y * 128;
  const int n0   = blockIdx.x * 128;

  v8f acc[2][4] = {};

  for (int k0 = 0; k0 < K; k0 += 32) {
    __syncthreads();
    {  // A tile 128x32: async b128 copies
      const int r = tid >> 1, h = (tid & 1) * 16;
      const _Float16* src = A + (size_t)(m0 + r) * K + k0 + h;
      copy16_g2l(src, &As[r][h]);
      copy16_g2l(src + 8, &As[r][h + 8]);
    }
    {  // B tile 32x128 -> transpose into Bt[n][k]
      const int kr = tid >> 3, c0 = (tid & 7) * 16;
      const _Float16* src = B + (size_t)(k0 + kr) * N + n0 + c0;
      v8h w0 = *(const v8h*)src;
      v8h w1 = *(const v8h*)(src + 8);
#pragma unroll
      for (int j = 0; j < 8; ++j) Bt[c0 + j][kr] = w0[j];
#pragma unroll
      for (int j = 0; j < 8; ++j) Bt[c0 + 8 + j][kr] = w1[j];
    }
    copy16_wait();
    __syncthreads();

    v16h af[2], bf[4];
#pragma unroll
    for (int i = 0; i < 2; ++i) af[i] = load_a_frag(&As[wm * 32 + i * 16][0], 40, lane);
#pragma unroll
    for (int j = 0; j < 4; ++j) bf[j] = load_b_frag_t(&Bt[wn * 64 + j * 16][0], 40, lane);
#pragma unroll
    for (int mi = 0; mi < 2; ++mi)
#pragma unroll
      for (int ni = 0; ni < 4; ++ni) acc[mi][ni] = wmma16(af[mi], bf[ni], acc[mi][ni]);
  }

  const int hi = lane >> 4, nn = lane & 15;
  if (Cf) {
#pragma unroll
    for (int mi = 0; mi < 2; ++mi)
#pragma unroll
      for (int ni = 0; ni < 4; ++ni) {
        const int mrow = m0 + wm * 32 + mi * 16 + hi * 8;
        const int ncol = n0 + wn * 64 + ni * 16 + nn;
        const float bv = bias[ncol];
#pragma unroll
        for (int r = 0; r < 8; ++r)
          Cf[(size_t)(mrow + r) * N + ncol] = acc[mi][ni][r] + bv;
      }
  }
  if (Ch) {
#pragma unroll
    for (int mi = 0; mi < 2; ++mi)
#pragma unroll
      for (int ni = 0; ni < 4; ++ni) {
        const int mrow = m0 + wm * 32 + mi * 16 + hi * 8;
        const int ncol = n0 + wn * 64 + ni * 16 + nn;
        const float bv = bias[ncol];
#pragma unroll
        for (int r = 0; r < 8; ++r)
          Ch[(size_t)(mrow + r) * N + ncol] = (_Float16)(acc[mi][ni][r] + bv);
      }
  }
}

// depthwise 3x3 conv, NHWC q[8,32,32,512] -> skip, zero pad, +bias
__global__ __launch_bounds__(256) void dwconv3x3_k(const float* __restrict__ q,
                                                   const float* __restrict__ w,
                                                   const float* __restrict__ bias,
                                                   float* __restrict__ skip) {
  const int idx = blockIdx.x * blockDim.x + threadIdx.x;  // 4194304
  const int c = idx & 511;
  const int x = (idx >> 9) & 31;
  const int y = (idx >> 14) & 31;
  const int b = idx >> 19;
  float acc = bias[c];
#pragma unroll
  for (int dy = 0; dy < 3; ++dy) {
    const int yy = y + dy - 1;
    if (yy < 0 || yy > 31) continue;
#pragma unroll
    for (int dx = 0; dx < 3; ++dx) {
      const int xx = x + dx - 1;
      if (xx < 0 || xx > 31) continue;
      acc += q[(size_t)(((b * 32 + yy) * 32 + xx)) * 512 + c] * w[c * 9 + dy * 3 + dx];
    }
  }
  skip[idx] = acc;
}

// fused flash attention: block = (b, h, 128-row q tile), 8 waves, wave = 16 q rows.
// K row-major (async), V transposed at stage time -> all frag loads contiguous.
__global__ __launch_bounds__(256) void attn_k(const _Float16* __restrict__ Q,
                                              const _Float16* __restrict__ KV,
                                              const float* __restrict__ skip,
                                              float* __restrict__ out) {
  __shared__ _Float16 Qs[128][80];     // q rows, row-major
  __shared__ _Float16 Ks[64][80];      // key rows, row-major (B^T via n-major read)
  __shared__ _Float16 Vt[64][40];      // Vt[c][key]
  __shared__ _Float16 Ps[8][16][80];   // per-wave P scratch

  const int b = blockIdx.z, h = blockIdx.y, qt = blockIdx.x;
  const int tid = threadIdx.x, lane = tid & 31, wave = tid >> 5;
  const int q0 = qt * 128;
  const int hi = lane >> 4, nn = lane & 15;

  for (int i = tid; i < 128 * 8; i += 256) {  // Q tile [128 x 64]
    const int r = i >> 3, cc = (i & 7) * 8;
    copy16_g2l(Q + ((size_t)(b * 1024 + q0 + r)) * 512 + h * 64 + cc, &Qs[r][cc]);
  }

  float mi[8], li[8];
  v8f o[4] = {};
#pragma unroll
  for (int r = 0; r < 8; ++r) { mi[r] = -1e30f; li[r] = 0.f; }
  const float scale = 0.125f;  // 1/sqrt(64)

  for (int n0 = 0; n0 < 1024; n0 += 64) {
    __syncthreads();  // previous chunk's readers done
    for (int i = tid; i < 64 * 8; i += 256) {
      const int r = i >> 3, cc = (i & 7) * 8;
      const _Float16* kk = KV + ((size_t)(b * 1024 + n0 + r)) * 1024 + h * 64 + cc;
      copy16_g2l(kk, &Ks[r][cc]);            // K rows: async
      v8h vv = *(const v8h*)(kk + 512);      // V: transpose into Vt[c][key]
#pragma unroll
      for (int j = 0; j < 8; ++j) Vt[cc + j][r] = vv[j];
    }
    copy16_wait();
    __syncthreads();

    // S[16x64] = Q_wave[16x64] * K^T
    v8f s[4] = {};
#pragma unroll
    for (int kt = 0; kt < 2; ++kt) {
      v16h a = load_a_frag(&Qs[wave * 16][kt * 32], 80, lane);
#pragma unroll
      for (int j = 0; j < 4; ++j)
        s[j] = wmma16(a, load_b_frag_t(&Ks[j * 16][kt * 32], 80, lane), s[j]);
    }

    // online softmax; C-layout rows: row = r + 8*hi, col = j*16 + nn
    float mnew[8];
#pragma unroll
    for (int j = 0; j < 4; ++j)
#pragma unroll
      for (int r = 0; r < 8; ++r) s[j][r] *= scale;
#pragma unroll
    for (int r = 0; r < 8; ++r)
      mnew[r] = fmaxf(fmaxf(s[0][r], s[1][r]), fmaxf(s[2][r], s[3][r]));
#pragma unroll
    for (int m = 1; m < 16; m <<= 1)
#pragma unroll
      for (int r = 0; r < 8; ++r) mnew[r] = fmaxf(mnew[r], __shfl_xor(mnew[r], m, 32));

    float rowsum[8];
#pragma unroll
    for (int r = 0; r < 8; ++r) {
      const float mn = fmaxf(mi[r], mnew[r]);
      const float corr = __expf(mi[r] - mn);
      mi[r] = mn;
      li[r] *= corr;
#pragma unroll
      for (int ct = 0; ct < 4; ++ct) o[ct][r] *= corr;
      rowsum[r] = 0.f;
    }
#pragma unroll
    for (int j = 0; j < 4; ++j)
#pragma unroll
      for (int r = 0; r < 8; ++r) {
        const float p = __expf(s[j][r] - mi[r]);
        rowsum[r] += p;
        Ps[wave][hi * 8 + r][j * 16 + nn] = (_Float16)p;  // wave-private
      }
#pragma unroll
    for (int m = 1; m < 16; m <<= 1)
#pragma unroll
      for (int r = 0; r < 8; ++r) rowsum[r] += __shfl_xor(rowsum[r], m, 32);
#pragma unroll
    for (int r = 0; r < 8; ++r) li[r] += rowsum[r];

    // O += P[16x64] * V[64x64]   (B(k=key, n=c) = Vt[c][key] -> contiguous)
#pragma unroll
    for (int kt = 0; kt < 2; ++kt) {
      v16h ap = load_a_frag(&Ps[wave][0][kt * 32], 80, lane);
#pragma unroll
      for (int ct = 0; ct < 4; ++ct)
        o[ct] = wmma16(ap, load_b_frag_t(&Vt[ct * 16][kt * 32], 40, lane), o[ct]);
    }
  }

  // epilogue: divide by row sum, add depthwise-conv skip, write f32
#pragma unroll
  for (int r = 0; r < 8; ++r) {
    const int m = q0 + wave * 16 + hi * 8 + r;
    const float inv = 1.f / li[r];
#pragma unroll
    for (int ct = 0; ct < 4; ++ct) {
      const int cc = h * 64 + ct * 16 + nn;
      const size_t idx = ((size_t)(b * 1024 + m)) * 512 + cc;
      out[idx] = o[ct][r] * inv + skip[idx];
    }
  }
}

// ------------------------------- launcher -------------------------------

extern "C" void kernel_launch(void* const* d_in, const int* in_sizes, int n_in,
                              void* d_out, int out_size, void* d_ws, size_t ws_size,
                              hipStream_t stream) {
  const float* query = (const float*)d_in[0];
  const float* key   = (const float*)d_in[1];
  const float* gq    = (const float*)d_in[2];
  const float* bqln  = (const float*)d_in[3];
  const float* gk    = (const float*)d_in[4];
  const float* bkln  = (const float*)d_in[5];
  const float* Wq    = (const float*)d_in[6];
  const float* bq    = (const float*)d_in[7];
  const float* Wkv   = (const float*)d_in[8];
  const float* bkv   = (const float*)d_in[9];
  const float* convw = (const float*)d_in[10];
  const float* convb = (const float*)d_in[11];
  float* out = (float*)d_out;
  (void)in_sizes; (void)n_in; (void)out_size; (void)ws_size;

  char* ws = (char*)d_ws;
  size_t off = 0;
  auto take = [&](size_t bytes) -> char* {
    char* p = ws + off;
    off += (bytes + 255) & ~(size_t)255;
    return p;
  };
  _Float16* lnq  = (_Float16*)take((size_t)8192 * 512 * 2);
  _Float16* lnk  = (_Float16*)take((size_t)8192 * 512 * 2);
  _Float16* wqh  = (_Float16*)take((size_t)512 * 512 * 2);
  _Float16* wkvh = (_Float16*)take((size_t)512 * 1024 * 2);
  float*    qf   = (float*)   take((size_t)8192 * 512 * 4);
  _Float16* qh   = (_Float16*)take((size_t)8192 * 512 * 2);
  _Float16* kvh  = (_Float16*)take((size_t)8192 * 1024 * 2);
  float*    skp  = (float*)   take((size_t)8192 * 512 * 4);

  cvt_f32_f16_k<<<dim3((512 * 512 + 255) / 256), dim3(256), 0, stream>>>(Wq, wqh, 512 * 512);
  cvt_f32_f16_k<<<dim3((512 * 1024 + 255) / 256), dim3(256), 0, stream>>>(Wkv, wkvh, 512 * 1024);
  ln512_f16_k<<<dim3(1024), dim3(256), 0, stream>>>(query, gq, bqln, lnq, 8192);
  ln512_f16_k<<<dim3(1024), dim3(256), 0, stream>>>(key, gk, bkln, lnk, 8192);
  // q = LN(query) @ Wq + bq : M=8192 N=512 K=512 (f32 for conv, f16 for attn)
  gemm_f16_k<<<dim3(512 / 128, 8192 / 128), dim3(256), 0, stream>>>(lnq, wqh, bq, qf, qh,
                                                                    8192, 512, 512);
  // kv = LN(key) @ Wkv + bkv : M=8192 N=1024 K=512 (f16 only)
  gemm_f16_k<<<dim3(1024 / 128, 8192 / 128), dim3(256), 0, stream>>>(lnk, wkvh, bkv, nullptr,
                                                                     kvh, 8192, 1024, 512);
  dwconv3x3_k<<<dim3(8 * 32 * 32 * 512 / 256), dim3(256), 0, stream>>>(qf, convw, convb, skp);
  attn_k<<<dim3(8, 8, 8), dim3(256), 0, stream>>>(qh, kvh, skp, out);
}
